// AdjustedNonLocalBlock_17970143167235
// MI455X (gfx1250) — matrix operand order repaired and checked by reading, not compiled, not run on hardware
//
#include <hip/hip_runtime.h>
#include <hip/hip_bf16.h>

typedef _Float16 half8 __attribute__((ext_vector_type(8)));
typedef _Float16 v16h  __attribute__((ext_vector_type(16)));
typedef float    v8f   __attribute__((ext_vector_type(8)));

#define BD  4
#define CD  128
#define CID 64
#define ND  4096   // H*W = 64*64

#define LOG2E 1.44269504088896340736f

// ---------------------------------------------------------------------------
// 16-lane max reduction in 4 steps via DPP source modifiers (folds into
// v_max_num_f32_dpp; no LDS, no permlane tied-movs).
//  0xB1 = quad_perm[1,0,3,2] (xor1), 0x4E = quad_perm[2,3,0,1] (xor2),
//  0x140 = row_mirror (quad q <-> 3-q), 0x141 = row_half_mirror (q <-> q^1)
// ---------------------------------------------------------------------------
template <int CTRL>
__device__ __forceinline__ float maxdpp(float x) {
    int xi = __builtin_bit_cast(int, x);
    int yi = __builtin_amdgcn_update_dpp(xi, xi, CTRL, 0xf, 0xf, false);
    return fmaxf(x, __builtin_bit_cast(float, yi));
}
__device__ __forceinline__ float rowmax16(float x) {
    x = maxdpp<0xB1>(x);
    x = maxdpp<0x4E>(x);
    x = maxdpp<0x140>(x);
    x = maxdpp<0x141>(x);
    return x;
}

// ---------------------------------------------------------------------------
// Kernel 1: three 1x1-conv projections (fp32 in, f16 out).
// theta pre-scaled by log2(e) so the softmax uses exp2 directly.
//   theta[b,n,ci] ([B,N,CI]), phiT[b,n,ci] ([B,N,CI]), gT[b,ci,n] ([B,CI,N])
// ---------------------------------------------------------------------------
__global__ void __launch_bounds__(256)
proj_kernel(const float* __restrict__ x0, const float* __restrict__ x1,
            const float* __restrict__ gw, const float* __restrict__ gb,
            const float* __restrict__ tw, const float* __restrict__ tb,
            const float* __restrict__ pw, const float* __restrict__ pb,
            _Float16* __restrict__ theta, _Float16* __restrict__ phiT,
            _Float16* __restrict__ gT)
{
    const int nchunks = ND / 256;
    int nc = blockIdx.x % nchunks;
    int ci = (blockIdx.x / nchunks) % CID;
    int b  = blockIdx.x / (nchunks * CID);
    int n  = nc * 256 + threadIdx.x;

    const float* x0p = x0 + (size_t)b * CD * ND + n;
    const float* x1p = x1 + (size_t)b * CD * ND + n;

    float at = tb[ci], ap = pb[ci], ag = gb[ci];
#pragma unroll 4
    for (int c = 0; c < CD; ++c) {
        float v0 = x0p[(size_t)c * ND];
        float v1 = x1p[(size_t)c * ND];
        at = fmaf(v1, tw[ci * CD + c], at);
        ap = fmaf(v0, pw[ci * CD + c], ap);
        ag = fmaf(v0, gw[ci * CD + c], ag);
    }
    theta[((size_t)b * ND + n) * CID + ci] = (_Float16)(at * LOG2E);
    phiT [((size_t)b * ND + n) * CID + ci] = (_Float16)ap;
    gT   [((size_t)b * CID + ci) * ND + n] = (_Float16)ag;
}

// ---------------------------------------------------------------------------
// Kernel 2: fused flash-attention over N=4096 keys.
// One wave32 = one 16-query block; key loop in blocks of 128.
//  S = Q K^T : 16x v_wmma_f32_16x16x32_f16  (logits in log2 domain)
//  row max   : DPP-folded v_max reduction (amortized over 128 keys)
//  row sum   : ones-augmented V -> 4 WMMAs accumulate Osum = P*1
//  O += P V  : 16x v_wmma_f32_16x16x32_f16
// Key->column permutation (tile kt, lane ln -> key 8*ln+kt) makes each lane's
// per-row P values 8 contiguous f16 -> one b128 LDS store per row.
// ---------------------------------------------------------------------------
__global__ void __launch_bounds__(128)
attn_kernel(const _Float16* __restrict__ theta, const _Float16* __restrict__ phiT,
            const _Float16* __restrict__ gT, float* __restrict__ y)
{
    __shared__ __attribute__((aligned(16))) _Float16 Plds[4][16][128]; // per-wave P staging

    const int lane   = threadIdx.x & 31;
    const int wid    = threadIdx.x >> 5;
    const int halfId = lane >> 4;     // 0 or 1
    const int ln     = lane & 15;

    const int qb = blockIdx.x * 4 + wid;   // global 16-query block id
    const int b  = qb >> 8;                // 256 q-blocks per batch
    const int q0 = (qb & 255) << 4;

    const _Float16* thB = theta + (size_t)b * ND * CID;
    const _Float16* phB = phiT  + (size_t)b * ND * CID;
    const _Float16* gB  = gT    + (size_t)b * CID * ND;

    union V16 { v16h v; half8 h[2]; };

    // --- Q fragments (A layout), two K=32 chunks covering CI=64 ---
    V16 qa[2];
    {
        const _Float16* qrow = thB + (size_t)(q0 + ln) * CID;
        const int koff = halfId * 16;
#pragma unroll
        for (int ch = 0; ch < 2; ++ch) {
            qa[ch].h[0] = *(const half8*)(qrow + ch * 32 + koff);
            qa[ch].h[1] = *(const half8*)(qrow + ch * 32 + 8 + koff);
        }
    }

    // all-ones B fragment for the row-sum WMMA
    v16h ones;
#pragma unroll
    for (int e = 0; e < 16; ++e) ones[e] = (_Float16)1.0f;

    v8f O[4], Osum;
#pragma unroll
    for (int t = 0; t < 4; ++t) O[t] = {};
    Osum = {};
    float mrow[8];
#pragma unroll
    for (int j = 0; j < 8; ++j) mrow[j] = -1e30f;

    for (int kb = 0; kb < ND / 128; ++kb) {
        // ---- S = Q K^T ; tile kt / lane ln covers key kb*128 + 8*ln + kt ----
        v8f S[8];
#pragma unroll
        for (int kt = 0; kt < 8; ++kt) {
            const _Float16* krow = phB + (size_t)(kb * 128 + 8 * ln + kt) * CID;
            v16h b0 = *(const v16h*)(krow + 0  + halfId * 16);
            v16h b1 = *(const v16h*)(krow + 32 + halfId * 16);
            v8f acc = {};
            acc = __builtin_amdgcn_wmma_f32_16x16x32_f16(false, qa[0].v, false, b0,
                                                         (short)0, acc, false, false);
            acc = __builtin_amdgcn_wmma_f32_16x16x32_f16(false, qa[1].v, false, b1,
                                                         (short)0, acc, false, false);
            S[kt] = acc;
        }

        // ---- online softmax (log2 domain); rows = halfId*8+j ----
#pragma unroll
        for (int j = 0; j < 8; ++j) {
            float m0 = fmaxf(fmaxf(S[0][j], S[1][j]), fmaxf(S[2][j], S[3][j]));
            float m1 = fmaxf(fmaxf(S[4][j], S[5][j]), fmaxf(S[6][j], S[7][j]));
            float mx = rowmax16(fmaxf(m0, m1));
            float mnew  = fmaxf(mrow[j], mx);
            float scale = __builtin_amdgcn_exp2f(mrow[j] - mnew);
            mrow[j] = mnew;
            // P in f16, packed: cols 8*ln .. 8*ln+7 of row halfId*8+j
            half8 pk;
#pragma unroll
            for (int kt = 0; kt < 8; ++kt)
                pk[kt] = (_Float16)__builtin_amdgcn_exp2f(S[kt][j] - mnew);
            *(half8*)&Plds[wid][halfId * 8 + j][8 * ln] = pk;
#pragma unroll
            for (int t = 0; t < 4; ++t) O[t][j] = O[t][j] * scale;
            Osum[j] = Osum[j] * scale;
        }
        // same-wave LDS ops are in-order; loads below see the stores above.

        // ---- O += P V ; Osum += P * ones ----
#pragma unroll
        for (int ch = 0; ch < 4; ++ch) {
            V16 pa;
            const _Float16* prow = &Plds[wid][ln][0];
            const int koff = halfId * 16;
            pa.h[0] = *(const half8*)(prow + ch * 32 + koff);
            pa.h[1] = *(const half8*)(prow + ch * 32 + 8 + koff);
#pragma unroll
            for (int t = 0; t < 4; ++t) {
                const _Float16* vrow =
                    gB + (size_t)(t * 16 + ln) * ND + kb * 128 + ch * 32 + halfId * 16;
                v16h bv = *(const v16h*)vrow;
                O[t] = __builtin_amdgcn_wmma_f32_16x16x32_f16(false, pa.v, false, bv,
                                                              (short)0, O[t], false, false);
            }
            Osum = __builtin_amdgcn_wmma_f32_16x16x32_f16(false, pa.v, false, ones,
                                                          (short)0, Osum, false, false);
        }
    }

    // ---- finalize: y[b, q, c] = O / rowsum ----
#pragma unroll
    for (int j = 0; j < 8; ++j) {
        float inv = 1.0f / Osum[j];      // all 16 columns of Osum hold the row sum
        int q = q0 + halfId * 8 + j;
        float* yrow = y + ((size_t)b * ND + q) * CID;
#pragma unroll
        for (int t = 0; t < 4; ++t)
            yrow[t * 16 + ln] = O[t][j] * inv;
    }
}

// ---------------------------------------------------------------------------
// Kernel 3: out[b,co,n] = W_b[co] + x0[b,co,n] + sum_ci W_w[co,ci]*y[b,n,ci]
// ---------------------------------------------------------------------------
__global__ void __launch_bounds__(256)
outconv_kernel(const float* __restrict__ y, const float* __restrict__ Ww,
               const float* __restrict__ Wb, const float* __restrict__ x0,
               float* __restrict__ out)
{
    const int nchunks = ND / 256;
    int nc = blockIdx.x % nchunks;
    int co = (blockIdx.x / nchunks) % CD;
    int b  = blockIdx.x / (nchunks * CD);
    int n  = nc * 256 + threadIdx.x;

    const float* yrow = y + ((size_t)b * ND + n) * CID;
    float acc = Wb[co];
#pragma unroll 8
    for (int ci = 0; ci < CID; ++ci)
        acc = fmaf(Ww[co * CID + ci], yrow[ci], acc);

    size_t idx = ((size_t)b * CD + co) * ND + n;
    out[idx] = acc + x0[idx];
}

// ---------------------------------------------------------------------------
extern "C" void kernel_launch(void* const* d_in, const int* in_sizes, int n_in,
                              void* d_out, int out_size, void* d_ws, size_t ws_size,
                              hipStream_t stream)
{
    const float* x0 = (const float*)d_in[0];
    const float* x1 = (const float*)d_in[1];
    const float* gw = (const float*)d_in[2];
    const float* gb = (const float*)d_in[3];
    const float* tw = (const float*)d_in[4];
    const float* tb = (const float*)d_in[5];
    const float* pw = (const float*)d_in[6];
    const float* pb = (const float*)d_in[7];
    const float* Ww = (const float*)d_in[8];
    const float* Wb = (const float*)d_in[9];
    float* out = (float*)d_out;

    const size_t projElems = (size_t)BD * ND * CID;      // 1M elements each
    _Float16* theta = (_Float16*)d_ws;                   // 2 MB
    _Float16* phiT  = theta + projElems;                 // 2 MB
    _Float16* gT    = phiT + projElems;                  // 2 MB
    float*    y     = (float*)(gT + projElems);          // 4 MB

    // 1) projections: grid B*CI*(N/256)
    proj_kernel<<<dim3(BD * CID * (ND / 256)), dim3(256), 0, stream>>>(
        x0, x1, gw, gb, tw, tb, pw, pb, theta, phiT, gT);

    // 2) fused flash attention: 4 waves/block, one 16-query block per wave
    attn_kernel<<<dim3(BD * (ND / 16) / 4), dim3(128), 0, stream>>>(
        theta, phiT, gT, y);

    // 3) output conv + bias + residual
    outconv_kernel<<<dim3(BD * CD * (ND / 256)), dim3(256), 0, stream>>>(
        y, Ww, Wb, x0, out);
}
